// CustomLSTM_18811956757257
// MI455X (gfx1250) — matrix operand order repaired and checked by reading, not compiled, not run on hardware
//
#include <hip/hip_runtime.h>
#include <hip/hip_bf16.h>

// ---------------------------------------------------------------------------
// Char-LSTM for MI455X (gfx1250, wave32, WMMA). Bandwidth-roofline design:
//  * Recurrent weights -> bf16, pre-swizzled into the native v_wmma 16x16x32
//    A-operand layout: 128 MB, resident in the 192 MB L2 (f32 would be 256 MB
//    -> HBM-bound, ~45 ms floor; bf16/L2 is several times faster).
//  * ONE persistent kernel runs all 4096 steps with a SINGLE device-wide
//    barrier per step. 256 blocks x 256 thr = 2048 waves, split-K x2:
//    each wave owns 16 rows of one gate over half of K (64 WMMA/step),
//    h broadcast into all 16 B columns (matvec is BW-bound; compute waste free).
//  * Cell update (phase 2) is computed REDUNDANTLY by every block into its own
//    LDS copy of h, with c carried in registers -> no second grid barrier.
//  * preact double-buffered (ping-pong on t&1) to decouple phase1(t+1) writes
//    from phase2(t) reads across blocks.
//  * one-hot column + bias folded into a [4][256][4096] table, applied in
//    phase 1 by the sk=0 wave. Decoder hs @ W_dec^T is a real WMMA GEMM.
// ---------------------------------------------------------------------------

typedef __attribute__((ext_vector_type(16))) __bf16 v16bf;
typedef __attribute__((ext_vector_type(8)))  float  v8f;

#define HIDDEN  4096
#define DCOLS   4352      // HIDDEN + VOCAB
#define VOCAB   256
#define SEQLEN  4096
#define KTILES  128       // 4096 / 32
#define KT_PER  64        // k-tiles per wave (split-K x2)
#define MTILES  256       // 4096 / 16
#define NBLK    256       // persistent blocks (must all be co-resident)
#define NTHR    256       // 8 waves per block

// ---- workspace layout (bytes) ----
static const size_t SZ_PW   = (size_t)4 * MTILES * KTILES * 512 * 2; // 128 MB packed gate weights (bf16)
static const size_t SZ_XB   = (size_t)4 * VOCAB * HIDDEN * 4;        // 16 MB  one-hot column + bias (f32)
static const size_t SZ_WDEC = (size_t)16 * KTILES * 512 * 2;         // 2 MB   packed decoder B (bf16)
static const size_t SZ_HS   = (size_t)SEQLEN * HIDDEN * 2;           // 32 MB  all hidden states (bf16)
static const size_t SZ_PRE  = (size_t)2 * 4 * HIDDEN * 4;            // 128 KB double-buffered preacts

static const size_t OFF_PW   = 0;
static const size_t OFF_XB   = OFF_PW   + SZ_PW;
static const size_t OFF_WDEC = OFF_XB   + SZ_XB;
static const size_t OFF_HS   = OFF_WDEC + SZ_WDEC;
static const size_t OFF_PRE  = OFF_HS   + SZ_HS;
static const size_t OFF_BAR  = OFF_PRE  + SZ_PRE;   // 2 x u32

// ---------------------------------------------------------------------------
// Pack recurrent weights W[:, :4096] (f32 row-major) into bf16 tiles laid out
// exactly as the 16x32 bf16 WMMA A operand (ISA 7.12.2):
//   lane 0-15 : M = lane,      lane 16-31 : M = lane-16 (K offset +8)
//   vgpr v (=j/2), half h (=j%2): K_local = (v<4?0:16) + grp*8 + (v&3)*2 + h
// Packed element index = (((gate*256+mt)*128)+kt)*512 + lane*16 + j
// -> each lane's 16 halves are 32 contiguous bytes; a wave's tile is 1 KB.
// ---------------------------------------------------------------------------
__global__ void k_pack_w(const float* __restrict__ Wf, const float* __restrict__ Wi,
                         const float* __restrict__ Wo, const float* __restrict__ Wg,
                         __bf16* __restrict__ pw) {
    unsigned tid = blockIdx.x * blockDim.x + threadIdx.x;   // < 4*256*128*512 = 67108864
    unsigned j    = tid & 15u;
    unsigned lane = (tid >> 4) & 31u;
    unsigned kt   = (tid >> 9) & 127u;
    unsigned mt   = (tid >> 16) & 255u;
    unsigned gate = tid >> 24;
    const float* W = (gate == 0) ? Wf : (gate == 1) ? Wi : (gate == 2) ? Wo : Wg;
    unsigned v = j >> 1, h = j & 1u, grp = lane >> 4;
    unsigned kl  = ((v < 4u) ? 0u : 16u) + grp * 8u + (v & 3u) * 2u + h;
    unsigned row = mt * 16u + (lane & 15u);
    unsigned col = kt * 32u + kl;
    pw[tid] = (__bf16)W[(size_t)row * DCOLS + col];
}

// xb[gate][char][r] = W_gate[r, 4096+char] + b_gate[r]   (f32)
__global__ void k_pack_xb(const float* __restrict__ Wf, const float* __restrict__ Wi,
                          const float* __restrict__ Wo, const float* __restrict__ Wg,
                          const float* __restrict__ bf_, const float* __restrict__ bi,
                          const float* __restrict__ bo, const float* __restrict__ bg,
                          float* __restrict__ xb) {
    unsigned tid = blockIdx.x * blockDim.x + threadIdx.x;   // < 4*256*4096 = 4194304
    unsigned r    = tid & 4095u;
    unsigned ch   = (tid >> 12) & 255u;
    unsigned gate = tid >> 20;
    const float* W = (gate == 0) ? Wf : (gate == 1) ? Wi : (gate == 2) ? Wo : Wg;
    const float* b = (gate == 0) ? bf_ : (gate == 1) ? bi : (gate == 2) ? bo : bg;
    xb[tid] = W[(size_t)r * DCOLS + HIDDEN + ch] + b[r];
}

// Pack W_dec [256,4096] f32 into bf16 WMMA B-operand layout:
//   B tile 32(K)x16(N): N_local = lane&15, K_local = (j<8?0:16) + (lane>>4)*8 + (j&7).
__global__ void k_pack_wdec(const float* __restrict__ Wdec, __bf16* __restrict__ pwd) {
    unsigned tid = blockIdx.x * blockDim.x + threadIdx.x;   // < 16*128*512 = 1048576
    unsigned j    = tid & 15u;
    unsigned lane = (tid >> 4) & 31u;
    unsigned kt   = (tid >> 9) & 127u;
    unsigned nt   = tid >> 16;
    pwd[tid] = (__bf16)Wdec[(size_t)(nt * 16u + (lane & 15u)) * HIDDEN + kt * 32u + ((j & 8u) << 1) + (lane >> 4) * 8u + (j & 7u)];
}

__global__ void k_init(unsigned* bar) {
    if (threadIdx.x < 2) bar[threadIdx.x] = 0u;
}

// Device-wide sense-reversing barrier (requires all NBLK blocks co-resident;
// 256 blocks x 8 waves x <9 KB LDS is far below any flagship's residency limit).
__device__ __forceinline__ void gsync(unsigned* bar) {
    __syncthreads();
    if (threadIdx.x == 0) {
        __threadfence();   // make this block's preact writes device-visible
        unsigned gen = __hip_atomic_load(bar + 1, __ATOMIC_RELAXED, __HIP_MEMORY_SCOPE_AGENT);
        unsigned a = __hip_atomic_fetch_add(bar, 1u, __ATOMIC_ACQ_REL, __HIP_MEMORY_SCOPE_AGENT);
        if (a == NBLK - 1) {
            __hip_atomic_store(bar, 0u, __ATOMIC_RELAXED, __HIP_MEMORY_SCOPE_AGENT);
            __hip_atomic_fetch_add(bar + 1, 1u, __ATOMIC_RELEASE, __HIP_MEMORY_SCOPE_AGENT);
        } else {
            while (__hip_atomic_load(bar + 1, __ATOMIC_ACQUIRE, __HIP_MEMORY_SCOPE_AGENT) == gen)
                __builtin_amdgcn_s_sleep(1);
        }
    }
    __syncthreads();
}

// ---------------------------------------------------------------------------
// Persistent LSTM kernel. 256 blocks x 256 threads = 2048 waves, split-K x2.
// Wave id = blk*8 + w: tile = id>>1 (gate = tile>>8, mt = tile&255), sk = id&1.
// Phase 1: 64 x v_wmma_f32_16x16x32_bf16 over K half [sk*64, sk*64+64);
//          split-K partials of a tile are reduced through LDS inside the
//          block; sk=0 wave adds the one-hot+bias term and writes preact.
// ONE grid barrier. Phase 2: every block redundantly updates all 4096 cells
// (c in registers, identical across blocks), writes h to its own LDS, and
// stores its 16-row slice of hs for the decoder.
// ---------------------------------------------------------------------------
__global__ void __launch_bounds__(NTHR, 1)
k_lstm(const __bf16* __restrict__ pw, const float* __restrict__ xb,
       const int* __restrict__ seq, float* __restrict__ pre,
       __bf16* __restrict__ hs, unsigned* bar) {
    __shared__ __bf16 lhb[HIDDEN];      // 8 KB: this block's copy of h (bf16)
    __shared__ float  part[8 * 16];     // split-K partial rows, per wave

    const int tid  = threadIdx.x;
    const int lane = tid & 31;
    const int w    = tid >> 5;                 // 0..7
    const int id   = blockIdx.x * 8 + w;       // 0..2047
    const int tile = id >> 1;                  // 0..1023
    const int sk   = id & 1;                   // split-K half
    const int gate = tile >> 8;
    const int mt   = tile & 255;
    const int grp  = lane >> 4;

    const __bf16* Abase = pw + ((size_t)tile * KTILES + (size_t)sk * KT_PER) * 512 + lane * 16;
    const unsigned short* lhu = (const unsigned short*)lhb;

    // init: h = 0 in LDS, c = 0 in registers (identical in every block)
    float c_reg[16];
    #pragma unroll
    for (int k = 0; k < 16; ++k) c_reg[k] = 0.0f;
    for (int i = tid; i < HIDDEN; i += NTHR) lhb[i] = (__bf16)0.0f;
    __syncthreads();

    for (int t = 0; t < SEQLEN; ++t) {
        // ---- phase 1: gate preactivations via WMMA (reads lhb) ----
        v8f acc;
        #pragma unroll
        for (int i = 0; i < 8; ++i) acc[i] = 0.0f;
        #pragma unroll 8
        for (int kt = 0; kt < KT_PER; ++kt) {
            v16bf a = *(const v16bf*)(Abase + (size_t)kt * 512);      // 32 B/lane contiguous
            union { uint4 q[2]; v16bf v; } b;
            b.q[0] = *(const uint4*)(lhu + (sk * KT_PER + kt) * 32 + grp * 8);
            b.q[1] = *(const uint4*)(lhu + (sk * KT_PER + kt) * 32 + 16 + grp * 8);
            acc = __builtin_amdgcn_wmma_f32_16x16x32_bf16(
                false, a, false, b.v, (short)0, acc, false, false);
        }
        // D: vgpr i -> M=i (lanes 0-15) / M=i+8 (lanes 16-31); columns equal,
        // so lanes 0 and 16 hold the 16 row results. Reduce split-K via LDS.
        if ((lane & 15) == 0) {
            #pragma unroll
            for (int i = 0; i < 8; ++i) part[w * 16 + grp * 8 + i] = acc[i];
        }
        __syncthreads();
        if (sk == 0 && (lane & 15) == 0) {
            int ch = seq[t];
            float* pp = pre + ((size_t)(t & 1) * 4 + gate) * HIDDEN + mt * 16 + grp * 8;
            const float* xp = xb + ((size_t)(gate * VOCAB + ch)) * HIDDEN + mt * 16 + grp * 8;
            #pragma unroll
            for (int i = 0; i < 8; ++i)
                pp[i] = part[w * 16 + grp * 8 + i] + part[(w ^ 1) * 16 + grp * 8 + i] + xp[i];
        }
        gsync(bar);   // the ONLY grid barrier per step

        // ---- phase 2: redundant full cell update in every block ----
        {
            const float* p0 = pre + (size_t)(t & 1) * 4 * HIDDEN;
            #pragma unroll 4
            for (int k = 0; k < 16; ++k) {
                int r = tid + NTHR * k;
                float pf = p0[0 * HIDDEN + r];
                float pi = p0[1 * HIDDEN + r];
                float po = p0[2 * HIDDEN + r];
                float pg = p0[3 * HIDDEN + r];
                float f  = 1.0f / (1.0f + __expf(-pf));
                float ii = 1.0f / (1.0f + __expf(-pi));
                float o  = 1.0f / (1.0f + __expf(-po));
                float g  = tanhf(pg);
                float c  = f * c_reg[k] + ii * g;
                c_reg[k] = c;
                float h  = o * tanhf(c);
                lhb[r]   = (__bf16)h;
            }
        }
        __syncthreads();   // lhb ready for next step's phase 1 (and hs slice)
        if (tid < 16)
            hs[(size_t)t * HIDDEN + blockIdx.x * 16 + tid] = lhb[blockIdx.x * 16 + tid];
    }
}

// ---------------------------------------------------------------------------
// Decoder: out[t, v] = hs[t, :] . W_dec[v, :] + b_dec[v]
// Real WMMA GEMM: M=4096 (time), N=256 (vocab), K=4096.
// One 16x16 tile per wave; 512 blocks x 8 waves = 4096 tiles.
// A built from row-major hs with two 16 B loads/lane matching the A layout.
// ---------------------------------------------------------------------------
__global__ void __launch_bounds__(256)
k_dec(const __bf16* __restrict__ hs, const __bf16* __restrict__ pwd,
      const float* __restrict__ bdec, float* __restrict__ out) {
    const int lane = threadIdx.x & 31;
    const int w    = threadIdx.x >> 5;
    const int tile = blockIdx.x * 8 + w;    // 0..4095
    const int mt   = tile >> 4;             // 0..255 time tiles
    const int nt   = tile & 15;             // 0..15 vocab tiles
    const int grp  = lane >> 4;
    const int nl   = lane & 15;

    v8f acc;
    #pragma unroll
    for (int i = 0; i < 8; ++i) acc[i] = 0.0f;

    const __bf16* Bbase = pwd + ((size_t)nt * KTILES) * 512 + lane * 16;
    const __bf16* Arow  = hs + (size_t)(mt * 16 + nl) * HIDDEN + grp * 8;

    #pragma unroll 8
    for (int kt = 0; kt < KTILES; ++kt) {
        union { uint4 q[2]; v16bf v; } a;
        a.q[0] = *(const uint4*)(Arow + kt * 32);       // K_local grp*8 + 0..7
        a.q[1] = *(const uint4*)(Arow + kt * 32 + 16);  // K_local 16+grp*8 + 0..7
        v16bf b = *(const v16bf*)(Bbase + (size_t)kt * 512);
        acc = __builtin_amdgcn_wmma_f32_16x16x32_bf16(
            false, a.v, false, b, (short)0, acc, false, false);
    }
    float bias = bdec[nt * 16 + nl];
    float* op = out + (size_t)(mt * 16 + grp * 8) * VOCAB + nt * 16 + nl;
    #pragma unroll
    for (int i = 0; i < 8; ++i) op[(size_t)i * VOCAB] = acc[i] + bias;
}

extern "C" void kernel_launch(void* const* d_in, const int* in_sizes, int n_in,
                              void* d_out, int out_size, void* d_ws, size_t ws_size,
                              hipStream_t stream) {
    const int*   seq  = (const int*)  d_in[0];
    const float* Wf   = (const float*)d_in[1];
    const float* bf_  = (const float*)d_in[2];
    const float* Wi   = (const float*)d_in[3];
    const float* bi   = (const float*)d_in[4];
    const float* Wo   = (const float*)d_in[5];
    const float* bo   = (const float*)d_in[6];
    const float* Wg   = (const float*)d_in[7];
    const float* bg   = (const float*)d_in[8];
    const float* Wdec = (const float*)d_in[9];
    const float* bdec = (const float*)d_in[10];
    float* out = (float*)d_out;

    char* ws = (char*)d_ws;
    __bf16*   pw   = (__bf16*)  (ws + OFF_PW);
    float*    xb   = (float*)   (ws + OFF_XB);
    __bf16*   pwd  = (__bf16*)  (ws + OFF_WDEC);
    __bf16*   hs   = (__bf16*)  (ws + OFF_HS);
    float*    pre  = (float*)   (ws + OFF_PRE);
    unsigned* bar  = (unsigned*)(ws + OFF_BAR);

    k_init<<<1, 64, 0, stream>>>(bar);
    k_pack_w<<<(4u * MTILES * KTILES * 512u) / 256u, 256, 0, stream>>>(Wf, Wi, Wo, Wg, pw);
    k_pack_xb<<<(4u * VOCAB * HIDDEN) / 256u, 256, 0, stream>>>(Wf, Wi, Wo, Wg, bf_, bi, bo, bg, xb);
    k_pack_wdec<<<(16u * KTILES * 512u) / 256u, 256, 0, stream>>>(Wdec, pwd);
    k_lstm<<<NBLK, NTHR, 0, stream>>>(pw, xb, seq, pre, hs, bar);
    k_dec<<<512, 256, 0, stream>>>(hs, pwd, bdec, out);
}
